// Block_32332513804635
// MI455X (gfx1250) — matrix operand order repaired and checked by reading, not compiled, not run on hardware
//
#include <hip/hip_runtime.h>
#include <cstdint>
#include <cstddef>

// ---------------------------------------------------------------------------
// MI455X (gfx1250) transformer block: LN -> ALiBi attention -> LN -> MoE.
// All heavy GEMMs use V_WMMA_F32_16X16X32_BF16 (fp32 accumulate), fp32->bf16
// conversion fused into the LDS tile load. Weights stream once (~660 MB) so
// we sit near the 23.3 TB/s HBM roofline instead of the fp32-FMA ceiling.
// Workspace requirement: ~170 MB.
// ---------------------------------------------------------------------------

typedef __attribute__((ext_vector_type(16))) __bf16 bf16x16;
typedef __attribute__((ext_vector_type(8)))  __bf16 bf16x8;
typedef __attribute__((ext_vector_type(8)))  float  f32x8;

// Problem dims (fixed by the reference)
constexpr int S  = 1024;
constexpr int D  = 2048;
constexpr int H  = 8;
constexpr int DK = 128;
constexpr int E  = 16;
constexpr int TOPK = 6;
constexpr int F  = 1408;
constexpr int SF = 2816;

// Epilogue modes
enum { EPI_NONE = 0, EPI_BIAS = 1, EPI_BIAS_RES = 2, EPI_MUL_SILU = 3, EPI_SCALE_ACC = 4 };

// GEMM tiling: 128x128 block tile, K-tile 64 (two wmma k-steps), 8 waves.
constexpr int BM = 128, BN = 128, BK = 64, SK = BK + 8;

union AFrag { bf16x16 v; bf16x8 h[2]; };

__device__ inline f32x8 zero8() { f32x8 z = {0.f,0.f,0.f,0.f,0.f,0.f,0.f,0.f}; return z; }
__device__ inline float silu_f(float x) { return x / (1.0f + __expf(-x)); }

// ---------------------------------------------------------------------------
// Generic WMMA GEMM: C[z] = epilogue( A[z] (TA? ^T) @ B[z] (TB? as [N][K]) )
// A,B fp32 in HBM, converted to bf16 in LDS. M,N multiples of 128, K of 64.
// ---------------------------------------------------------------------------
template<bool TA, bool TB, int EPI>
__global__ __launch_bounds__(256) void gemm_wmma(
    int M, int N, int K,
    const float* __restrict__ A, int lda, long sA,
    const float* __restrict__ B, int ldb, long sB,
    float*       __restrict__ C, int ldc, long sC,
    const float* __restrict__ bias, long sBias,
    const float* __restrict__ aux, int ldaux, long sAux,
    const float* __restrict__ rsc, int rsStride, long sRsc)
{
  (void)M; (void)N;
  __shared__ __bf16 As[BM * SK];
  __shared__ __bf16 Bs[BN * SK];

  const int tid  = threadIdx.x;
  const int lane = tid & 31;
  const int wave = tid >> 5;
  const int wm   = wave >> 2;        // 0..1  (64 rows of C each)
  const int wn   = wave & 3;         // 0..3  (32 cols of C each)
  const int hlf  = lane >> 4;        // wave32 half selector
  const int lr   = lane & 15;
  const int m0   = blockIdx.y * BM;
  const int n0   = blockIdx.x * BN;
  const long z   = blockIdx.z;

  A += z * sA; B += z * sB; C += z * sC;
  const float* biasp = bias ? bias + z * sBias : nullptr;
  const float* auxp  = aux  ? aux  + z * sAux  : nullptr;
  const float* rscp  = rsc  ? rsc  + z * sRsc  : nullptr;

  f32x8 acc[4][2];
#pragma unroll
  for (int mi = 0; mi < 4; ++mi)
#pragma unroll
    for (int ni = 0; ni < 2; ++ni) acc[mi][ni] = zero8();

  for (int kt = 0; kt < K; kt += BK) {
    // ---- A tile -> LDS As[m][k] (bf16) ----
    if constexpr (!TA) {
#pragma unroll
      for (int it = 0; it < 8; ++it) {
        const int idx = tid + it * 256;          // 128 rows x 16 float4
        const int m  = idx >> 4;
        const int kv = (idx & 15) << 2;
        const float4 f = *(const float4*)(A + (size_t)(m0 + m) * lda + kt + kv);
        __bf16* d = &As[m * SK + kv];
        d[0] = (__bf16)f.x; d[1] = (__bf16)f.y; d[2] = (__bf16)f.z; d[3] = (__bf16)f.w;
      }
      if (kt + BK < K) __builtin_prefetch(A + (size_t)(m0 + (tid >> 4)) * lda + kt + BK, 0, 0);
    } else {
#pragma unroll
      for (int it = 0; it < 8; ++it) {
        const int idx = tid + it * 256;          // 64 k-rows x 32 float4 (along m)
        const int mv = (idx & 31) << 2;
        const int k  = idx >> 5;
        const float4 f = *(const float4*)(A + (size_t)(kt + k) * lda + m0 + mv);
        As[(mv + 0) * SK + k] = (__bf16)f.x;
        As[(mv + 1) * SK + k] = (__bf16)f.y;
        As[(mv + 2) * SK + k] = (__bf16)f.z;
        As[(mv + 3) * SK + k] = (__bf16)f.w;
      }
    }
    // ---- B tile -> LDS Bs[n][k] (bf16, always [N][K] in LDS) ----
    if constexpr (!TB) {
#pragma unroll
      for (int it = 0; it < 8; ++it) {
        const int idx = tid + it * 256;          // 64 k-rows x 32 float4 (along n)
        const int nv = (idx & 31) << 2;
        const int k  = idx >> 5;
        const float4 f = *(const float4*)(B + (size_t)(kt + k) * ldb + n0 + nv);
        Bs[(nv + 0) * SK + k] = (__bf16)f.x;
        Bs[(nv + 1) * SK + k] = (__bf16)f.y;
        Bs[(nv + 2) * SK + k] = (__bf16)f.z;
        Bs[(nv + 3) * SK + k] = (__bf16)f.w;
      }
      if (kt + BK < K) __builtin_prefetch(B + (size_t)(kt + BK + (tid >> 5)) * ldb + n0, 0, 0);
    } else {
#pragma unroll
      for (int it = 0; it < 8; ++it) {
        const int idx = tid + it * 256;          // 128 n-rows x 16 float4 (along k)
        const int n  = idx >> 4;
        const int kv = (idx & 15) << 2;
        const float4 f = *(const float4*)(B + (size_t)(n0 + n) * ldb + kt + kv);
        __bf16* d = &Bs[n * SK + kv];
        d[0] = (__bf16)f.x; d[1] = (__bf16)f.y; d[2] = (__bf16)f.z; d[3] = (__bf16)f.w;
      }
      if (kt + BK < K) __builtin_prefetch(B + (size_t)(n0 + (tid >> 4)) * ldb + kt + BK, 0, 0);
    }
    __syncthreads();

    // ---- 2 x (8 wmma) per K-tile ----
#pragma unroll
    for (int kk = 0; kk < BK; kk += 32) {
      AFrag fa[4], fb[2];
#pragma unroll
      for (int mi = 0; mi < 4; ++mi) {
        // 16x32 bf16 A frag: lanes 0-15 hold K0..7 & K16..23; lanes 16-31 K8..15 & K24..31
        const __bf16* pa = &As[(wm * 64 + mi * 16 + lr) * SK + kk + hlf * 8];
        fa[mi].h[0] = *(const bf16x8*)pa;
        fa[mi].h[1] = *(const bf16x8*)(pa + 16);
      }
#pragma unroll
      for (int ni = 0; ni < 2; ++ni) {
        // 32x16 bf16 B frag: lanes 0-15 hold K0..7 & K16..23 of col n; lanes 16-31 K8..15 & K24..31
        const __bf16* pb = &Bs[(wn * 32 + ni * 16 + lr) * SK + kk + hlf * 8];
        fb[ni].h[0] = *(const bf16x8*)pb;
        fb[ni].h[1] = *(const bf16x8*)(pb + 16);
      }
#pragma unroll
      for (int mi = 0; mi < 4; ++mi)
#pragma unroll
        for (int ni = 0; ni < 2; ++ni)
          acc[mi][ni] = __builtin_amdgcn_wmma_f32_16x16x32_bf16(
              false, fa[mi].v, false, fb[ni].v, (short)0, acc[mi][ni], false, false);
    }
    __syncthreads();
  }

  // ---- epilogue: C/D layout: VGPR i -> M = i + (lane>=16 ? 8 : 0), N = lane&15 ----
#pragma unroll
  for (int mi = 0; mi < 4; ++mi) {
#pragma unroll
    for (int ni = 0; ni < 2; ++ni) {
      const int n = n0 + wn * 32 + ni * 16 + lr;
      float bn = 0.f;
      if constexpr (EPI != EPI_NONE) bn = biasp[n];
#pragma unroll
      for (int i = 0; i < 8; ++i) {
        const int m = m0 + wm * 64 + mi * 16 + i + hlf * 8;
        const float v = acc[mi][ni][i];
        const size_t ci = (size_t)m * ldc + n;
        if constexpr (EPI == EPI_NONE)           C[ci] = v;
        else if constexpr (EPI == EPI_BIAS)      C[ci] = v + bn;
        else if constexpr (EPI == EPI_BIAS_RES)  C[ci] = v + bn + auxp[(size_t)m * ldaux + n];
        else if constexpr (EPI == EPI_MUL_SILU)  C[ci] = silu_f(auxp[(size_t)m * ldaux + n]) * (v + bn);
        else if constexpr (EPI == EPI_SCALE_ACC) C[ci] += rscp[(size_t)m * rsStride] * (v + bn);
      }
    }
  }
}

// ---------------------------------------------------------------------------
// layer_norm (mean-subtracting, weight only), one block per row
// ---------------------------------------------------------------------------
__global__ __launch_bounds__(256) void ln_kernel(
    const float* __restrict__ x, const float* __restrict__ g, float* __restrict__ o)
{
  __shared__ float rs[256], rq[256];
  const int row = blockIdx.x;
  const float* xr = x + (size_t)row * D;
  float a = 0.f, b = 0.f;
  for (int d = threadIdx.x; d < D; d += 256) { const float v = xr[d]; a += v; b += v * v; }
  rs[threadIdx.x] = a; rq[threadIdx.x] = b;
  __syncthreads();
  for (int s2 = 128; s2 > 0; s2 >>= 1) {
    if (threadIdx.x < s2) { rs[threadIdx.x] += rs[threadIdx.x + s2]; rq[threadIdx.x] += rq[threadIdx.x + s2]; }
    __syncthreads();
  }
  const float mu  = rs[0] / (float)D;
  const float var = rq[0] / (float)D - mu * mu;
  const float ri  = rsqrtf(var + 1e-8f);
  float* orow = o + (size_t)row * D;
  for (int d = threadIdx.x; d < D; d += 256) orow[d] = (xr[d] - mu) * ri * g[d];
}

// ---------------------------------------------------------------------------
// Fused scale + ALiBi + causal-mask + softmax over each score row [h, i, :]
// ---------------------------------------------------------------------------
__global__ __launch_bounds__(256) void softmax_alibi_kernel(
    float* __restrict__ sc, const float* __restrict__ mask)
{
  __shared__ float red[256];
  const int i = blockIdx.x, h = blockIdx.y;
  float* row = sc + ((size_t)h * S + i) * S;
  const float* mrow = mask + (size_t)i * S;
  const float slope = exp2f(-(float)(h + 1));
  const float scale = 0.088388347648318447f;   // 1/sqrt(128)
  float vreg[4];
  float mx = -3.4e38f;
#pragma unroll
  for (int r = 0; r < 4; ++r) {
    const int j = threadIdx.x + r * 256;
    const float rel = (i >= j) ? -(float)(i - j) : 0.f;
    const float v = row[j] * scale + slope * rel + mrow[j];
    vreg[r] = v; mx = fmaxf(mx, v);
  }
  red[threadIdx.x] = mx; __syncthreads();
  for (int s2 = 128; s2 > 0; s2 >>= 1) {
    if (threadIdx.x < s2) red[threadIdx.x] = fmaxf(red[threadIdx.x], red[threadIdx.x + s2]);
    __syncthreads();
  }
  mx = red[0]; __syncthreads();
  float sum = 0.f;
#pragma unroll
  for (int r = 0; r < 4; ++r) { vreg[r] = __expf(vreg[r] - mx); sum += vreg[r]; }
  red[threadIdx.x] = sum; __syncthreads();
  for (int s2 = 128; s2 > 0; s2 >>= 1) {
    if (threadIdx.x < s2) red[threadIdx.x] += red[threadIdx.x + s2];
    __syncthreads();
  }
  const float inv = 1.0f / red[0];
#pragma unroll
  for (int r = 0; r < 4; ++r) row[threadIdx.x + r * 256] = vreg[r] * inv;
}

// ---------------------------------------------------------------------------
// Gate: logits = x @ gate_w^T, softmax(16), top-6 on (softmax + bias),
// dense combine weights wd[t, e] (softmax prob or 0). One block per token.
// ---------------------------------------------------------------------------
__global__ __launch_bounds__(256) void gate_topk_kernel(
    const float* __restrict__ xn, const float* __restrict__ gw,
    const float* __restrict__ gb, float* __restrict__ wd)
{
  __shared__ float red[E][17];
  __shared__ float lg[E];
  const int t = blockIdx.x;
  const int e = threadIdx.x >> 4;
  const int p = threadIdx.x & 15;
  const float* xr = xn + (size_t)t * D;
  const float* wr = gw + (size_t)e * D;
  float a = 0.f;
  for (int d = p; d < D; d += 16) a += xr[d] * wr[d];
  red[e][p] = a;
  __syncthreads();
  if (threadIdx.x < E) {
    float s2 = 0.f;
    for (int j = 0; j < 16; ++j) s2 += red[threadIdx.x][j];
    lg[threadIdx.x] = s2;
  }
  __syncthreads();
  if (threadIdx.x == 0) {
    float mx = lg[0];
    for (int i = 1; i < E; ++i) mx = fmaxf(mx, lg[i]);
    float sm[E], sel[E], out[E];
    float ssum = 0.f;
    for (int i = 0; i < E; ++i) { sm[i] = __expf(lg[i] - mx); ssum += sm[i]; }
    for (int i = 0; i < E; ++i) { sm[i] /= ssum; sel[i] = sm[i] + gb[i]; out[i] = 0.f; }
    for (int k = 0; k < TOPK; ++k) {
      int best = 0; float bv = sel[0];
      for (int i = 1; i < E; ++i) if (sel[i] > bv) { bv = sel[i]; best = i; }
      out[best] = sm[best];
      sel[best] = -3.4e38f;
    }
    for (int i = 0; i < E; ++i) wd[(size_t)t * E + i] = out[i];
  }
}

__global__ __launch_bounds__(256) void add_kernel(
    const float* __restrict__ a, const float* __restrict__ b, float* __restrict__ o, int n)
{
  const int idx = blockIdx.x * 256 + threadIdx.x;
  if (idx < n) o[idx] = a[idx] + b[idx];
}

// ---------------------------------------------------------------------------
extern "C" void kernel_launch(void* const* d_in, const int* in_sizes, int n_in,
                              void* d_out, int out_size, void* d_ws, size_t ws_size,
                              hipStream_t stream)
{
  (void)in_sizes; (void)n_in; (void)out_size; (void)ws_size;
  const float* x     = (const float*)d_in[0];
  const float* mask  = (const float*)d_in[1];
  const float* attng = (const float*)d_in[2];
  const float* wq_w  = (const float*)d_in[3];
  const float* wq_b  = (const float*)d_in[4];
  const float* wk_w  = (const float*)d_in[5];
  const float* wk_b  = (const float*)d_in[6];
  const float* wv_w  = (const float*)d_in[7];
  const float* wv_b  = (const float*)d_in[8];
  const float* wo_w  = (const float*)d_in[9];
  const float* wo_b  = (const float*)d_in[10];
  const float* ffng  = (const float*)d_in[11];
  const float* gw    = (const float*)d_in[12];
  const float* gb    = (const float*)d_in[13];
  const float* e_w1  = (const float*)d_in[14];
  const float* e_b1  = (const float*)d_in[15];
  const float* e_w2  = (const float*)d_in[16];
  const float* e_b2  = (const float*)d_in[17];
  const float* e_w3  = (const float*)d_in[18];
  const float* e_b3  = (const float*)d_in[19];
  const float* s_w1  = (const float*)d_in[20];
  const float* s_b1  = (const float*)d_in[21];
  const float* s_w2  = (const float*)d_in[22];
  const float* s_b2  = (const float*)d_in[23];
  const float* s_w3  = (const float*)d_in[24];
  const float* s_b3  = (const float*)d_in[25];
  // d_in[26] = start_pos (unused by the reference forward)

  float* ws  = (float*)d_ws;
  float* xn1 = ws; ws += (size_t)S * D;       // LN(x)
  float* q   = ws; ws += (size_t)S * H * DK;
  float* kb  = ws; ws += (size_t)S * H * DK;
  float* v   = ws; ws += (size_t)S * H * DK;
  float* sc  = ws; ws += (size_t)H * S * S;   // scores / probs (in place)
  float* ao  = ws; ws += (size_t)S * H * DK;  // p^T @ v
  float* h1  = ws; ws += (size_t)S * D;       // x + attn
  float* xn2 = ws; ws += (size_t)S * D;       // LN(h1)
  float* wd  = ws; ws += (size_t)S * E;       // dense routing weights
  float* y   = ws; ws += (size_t)S * D;       // MoE accumulator (shared + routed)
  float* hb  = ws; ws += (size_t)E * S * F;   // per-expert SwiGLU hidden (also shared)

  const dim3 blk(256);
  const int HD = H * DK;   // 1024

  // ---- attention ----
  ln_kernel<<<S, blk, 0, stream>>>(x, attng, xn1);

  gemm_wmma<false,false,EPI_BIAS><<<dim3(HD/BN, S/BM, 1), blk, 0, stream>>>(
      S, HD, D, xn1, D, 0, wq_w, HD, 0, q, HD, 0, wq_b, 0, nullptr, 0, 0, nullptr, 0, 0);
  gemm_wmma<false,false,EPI_BIAS><<<dim3(HD/BN, S/BM, 1), blk, 0, stream>>>(
      S, HD, D, xn1, D, 0, wk_w, HD, 0, kb, HD, 0, wk_b, 0, nullptr, 0, 0, nullptr, 0, 0);
  gemm_wmma<false,false,EPI_BIAS><<<dim3(HD/BN, S/BM, 1), blk, 0, stream>>>(
      S, HD, D, xn1, D, 0, wv_w, HD, 0, v, HD, 0, wv_b, 0, nullptr, 0, 0, nullptr, 0, 0);

  // scores[h] = q_h @ k_h^T  (B given as [N][K] -> TB)
  gemm_wmma<false,true,EPI_NONE><<<dim3(S/BN, S/BM, H), blk, 0, stream>>>(
      S, S, DK, q, HD, DK, kb, HD, DK, sc, S, (long)S*S, nullptr, 0, nullptr, 0, 0, nullptr, 0, 0);

  softmax_alibi_kernel<<<dim3(S, H), blk, 0, stream>>>(sc, mask);

  // out_h = p_h^T @ v_h   (faithful to einsum 'bhsS,bshv->bShv'; A transposed)
  gemm_wmma<true,false,EPI_NONE><<<dim3(DK/BN, S/BM, H), blk, 0, stream>>>(
      S, DK, S, sc, S, (long)S*S, v, HD, DK, ao, HD, DK, nullptr, 0, nullptr, 0, 0, nullptr, 0, 0);

  // h1 = x + (ao @ wo + wo_b)
  gemm_wmma<false,false,EPI_BIAS_RES><<<dim3(D/BN, S/BM, 1), blk, 0, stream>>>(
      S, D, HD, ao, HD, 0, wo_w, D, 0, h1, D, 0, wo_b, 0, x, D, 0, nullptr, 0, 0);

  // ---- MoE ----
  ln_kernel<<<S, blk, 0, stream>>>(h1, ffng, xn2);
  gate_topk_kernel<<<S, blk, 0, stream>>>(xn2, gw, gb, wd);

  // shared expert (uses front of hb as scratch): y = SwiGLU(xn2) @ s_w2 + s_b2
  gemm_wmma<false,false,EPI_BIAS><<<dim3(SF/BN, S/BM, 1), blk, 0, stream>>>(
      S, SF, D, xn2, D, 0, s_w1, SF, 0, hb, SF, 0, s_b1, 0, nullptr, 0, 0, nullptr, 0, 0);
  gemm_wmma<false,false,EPI_MUL_SILU><<<dim3(SF/BN, S/BM, 1), blk, 0, stream>>>(
      S, SF, D, xn2, D, 0, s_w3, SF, 0, hb, SF, 0, s_b3, 0, hb, SF, 0, nullptr, 0, 0);
  gemm_wmma<false,false,EPI_BIAS><<<dim3(D/BN, S/BM, 1), blk, 0, stream>>>(
      S, D, SF, hb, SF, 0, s_w2, D, 0, y, D, 0, s_b2, 0, nullptr, 0, 0, nullptr, 0, 0);

  // routed experts, SwiGLU hidden batched over all 16 experts (blockIdx.z = e)
  gemm_wmma<false,false,EPI_BIAS><<<dim3(F/BN, S/BM, E), blk, 0, stream>>>(
      S, F, D, xn2, D, 0, e_w1, F, (long)D*F, hb, F, (long)S*F, e_b1, F,
      nullptr, 0, 0, nullptr, 0, 0);
  gemm_wmma<false,false,EPI_MUL_SILU><<<dim3(F/BN, S/BM, E), blk, 0, stream>>>(
      S, F, D, xn2, D, 0, e_w3, F, (long)D*F, hb, F, (long)S*F, e_b3, F,
      hb, F, (long)S*F, nullptr, 0, 0);

  // down-projection + routed combine: y += wd[:,e] * (h_e @ w2_e + b2_e)
  // sequential per expert -> plain fp32 read-modify-write, no atomics
  for (int e = 0; e < E; ++e) {
    gemm_wmma<false,false,EPI_SCALE_ACC><<<dim3(D/BN, S/BM, 1), blk, 0, stream>>>(
        S, D, F, hb + (size_t)e * S * F, F, 0, e_w2 + (size_t)e * F * D, D, 0,
        y, D, 0, e_b2 + (size_t)e * D, 0, nullptr, 0, 0, wd + e, E, 0);
  }

  // ---- final residual: out = h1 + moe ----
  add_kernel<<<dim3((S * D) / 256), blk, 0, stream>>>(h1, y, (float*)d_out, S * D);
}
